// CRF_25984552141127
// MI455X (gfx1250) — matrix operand order repaired and checked by reading, not compiled, not run on hardware
//
#include <hip/hip_runtime.h>

typedef __attribute__((ext_vector_type(16))) _Float16 v16h;
typedef __attribute__((ext_vector_type(8)))  _Float16 v8h;
typedef __attribute__((ext_vector_type(8)))  float    v8f;

#define CRF_B 512
#define CRF_S 1024
#define CRF_L 64
#define T_BOS 61
#define T_EOS 62

#define LOG2E 1.44269504088896340736f
#define LN2   0.69314718055994530942f

// ---------------------------------------------------------------------------
// Kernel 0: expT = exp(trans) split into f16 hi/lo tables (64x64 each).
// exp(-10000) -> 0 encodes forbidden transitions exactly.
// ---------------------------------------------------------------------------
__global__ __launch_bounds__(256) void crf_prep(const float* __restrict__ trans,
                                                _Float16* __restrict__ eh,
                                                _Float16* __restrict__ el) {
    int idx = blockIdx.x * blockDim.x + threadIdx.x;
    if (idx < CRF_L * CRF_L) {
        float v = __builtin_amdgcn_exp2f(trans[idx] * LOG2E);
        _Float16 h = (_Float16)v;
        eh[idx] = h;
        el[idx] = (_Float16)(v - (float)h);
    }
}

// ---------------------------------------------------------------------------
// Kernel 1: gold path score, one wave32 per batch row.
// ---------------------------------------------------------------------------
__global__ __launch_bounds__(256) void crf_gold(const float* __restrict__ em,
                                                const float* __restrict__ trans,
                                                const int*   __restrict__ tags,
                                                const float* __restrict__ mask,
                                                float* __restrict__ gold) {
    int b    = blockIdx.x * (blockDim.x >> 5) + (threadIdx.x >> 5);
    int lane = threadIdx.x & 31;
    if (b >= CRF_B) return;

    const float* emb = em   + (size_t)b * CRF_S * CRF_L;
    const int*   tg  = tags + (size_t)b * CRF_S;
    const float* mk  = mask + (size_t)b * CRF_S;

    float acc = 0.0f, msum = 0.0f;
    for (int t = lane; t < CRF_S; t += 32) {
        float mt = mk[t];
        msum += mt;
        int tt = tg[t];
        if (t == 0) {
            acc += trans[T_BOS * CRF_L + tt] + emb[tt];
        } else {
            acc += (emb[(size_t)t * CRF_L + tt] + trans[tg[t - 1] * CRF_L + tt]) * mt;
        }
    }
    #pragma unroll
    for (int off = 16; off > 0; off >>= 1) {
        acc  += __shfl_xor(acc,  off);
        msum += __shfl_xor(msum, off);
    }
    if (lane == 0) {
        int last = (int)(msum + 0.5f) - 1;
        last = last < 0 ? 0 : (last >= CRF_S ? CRF_S - 1 : last);
        gold[b] = acc + trans[tg[last] * CRF_L + T_EOS];
    }
}

// ---------------------------------------------------------------------------
// Kernel 2: forward scan (log-partition) via WMMA, minus gold.
// One workgroup = 16 batch rows; wave mc owns output labels [16*mc, 16*mc+16).
// Per step: S^T = E^T x P^T with f16 hi/lo compensated products:
//   s = ph*Eh + pl*Eh + ph*El   (6x v_wmma_f32_16x16x32_f16, fp32 accum)
// Two barriers per step; exp/log use raw v_exp_f32 / v_log_f32.
// ---------------------------------------------------------------------------
__global__ __launch_bounds__(128) void crf_forward(const float* __restrict__ em,
                                                   const float* __restrict__ trans,
                                                   const float* __restrict__ mask,
                                                   const _Float16* __restrict__ eh,
                                                   const _Float16* __restrict__ el,
                                                   const float* __restrict__ gold,
                                                   float* __restrict__ out) {
    __shared__ float    alpha[16 * 68];   // padded rows: bank-conflict free
    __shared__ float    mrow[16];
    __shared__ _Float16 phh[16 * 72];     // p hi, padded rows
    __shared__ _Float16 pll[16 * 72];     // p lo

    const int tid  = threadIdx.x;
    const int lane = tid & 31;
    const int mc   = tid >> 5;            // wave id = label chunk
    const int gb0  = blockIdx.x * 16;

    const int bloc = tid >> 3;            // 0..15 (phases A/B: batch row)
    const int j0   = (tid & 7) * 8;       // 8 labels per thread

    const int n  = lane & 15;             // WMMA N = batch within tile
    const int hi = lane >> 4;             // lane half
    const int jb = mc * 16 + hi * 8;      // first label of this lane's D regs

    // ---- preload constant A fragments: A[m][k] = E^T[j][i] = exp(T[i][j]) ----
    // ISA 16-bit A 16x32 layout: M = lane&15;
    // K = (v<4 ? 2v : 16+2(v-4)) + half + (lane>=16 ? 8 : 0)
    v16h Eh0, Eh1, El0, El1;
    {
        const int j = mc * 16 + n;
        #pragma unroll
        for (int h = 0; h < 16; ++h) {
            int v = h >> 1;
            int K = ((v < 4) ? (2 * v) : (16 + 2 * (v - 4))) + (h & 1) + (hi ? 8 : 0);
            Eh0[h] = eh[K * CRF_L + j];
            El0[h] = el[K * CRF_L + j];
            Eh1[h] = eh[(K + 32) * CRF_L + j];
            El1[h] = el[(K + 32) * CRF_L + j];
        }
    }

    // ---- t = 0: alpha = trans[BOS, :] + em[:, 0, :] ----
    {
        const float* e0p = em + (size_t)(gb0 + bloc) * CRF_S * CRF_L;
        #pragma unroll
        for (int k = 0; k < 8; ++k)
            alpha[bloc * 68 + j0 + k] = trans[T_BOS * CRF_L + j0 + k] + e0p[j0 + k];
    }
    __syncthreads();

    // emissions / mask double buffers (addresses independent of recursion)
    const float* emL = em + (size_t)(gb0 + n) * CRF_S * CRF_L + jb;
    const float* mkL = mask + (size_t)(gb0 + n) * CRF_S;
    float4 e0 = *(const float4*)(emL + 1 * CRF_L);
    float4 e1 = *(const float4*)(emL + 1 * CRF_L + 4);
    float  mtc = mkL[1];

    for (int t = 1; t < CRF_S; ++t) {
        // ---- phase A: load alpha row chunk, row max via 8-lane shuffles ----
        float4 a0 = *(const float4*)&alpha[bloc * 68 + j0];
        float4 a1 = *(const float4*)&alpha[bloc * 68 + j0 + 4];
        float av[8] = {a0.x, a0.y, a0.z, a0.w, a1.x, a1.y, a1.z, a1.w};
        float pm = -3.402823466e38f;
        #pragma unroll
        for (int k = 0; k < 8; ++k) pm = fmaxf(pm, av[k]);
        pm = fmaxf(pm, __shfl_xor(pm, 1));
        pm = fmaxf(pm, __shfl_xor(pm, 2));
        pm = fmaxf(pm, __shfl_xor(pm, 4));
        if ((tid & 7) == 0) mrow[bloc] = pm;   // cross-wave consumers read after barrier

        // ---- phase B: p = exp(alpha - m) as f16 hi + lo (raw v_exp_f32) ----
        v8h hv, lv;
        #pragma unroll
        for (int k = 0; k < 8; ++k) {
            float v = __builtin_amdgcn_exp2f((av[k] - pm) * LOG2E);
            _Float16 h = (_Float16)v;
            hv[k] = h;
            lv[k] = (_Float16)(v - (float)h);
        }
        *(v8h*)&phh[bloc * 72 + j0] = hv;
        *(v8h*)&pll[bloc * 72 + j0] = lv;
        __syncthreads();

        // ---- phase C: WMMA + log + emission add ----
        float4 c0 = e0, c1 = e1;
        float  mt = mtc;
        if (t + 1 < CRF_S) {
            e0  = *(const float4*)(emL + (size_t)(t + 1) * CRF_L);
            e1  = *(const float4*)(emL + (size_t)(t + 1) * CRF_L + 4);
            mtc = mkL[t + 1];
        }
        if (t + 8 < CRF_S) __builtin_prefetch(emL + (size_t)(t + 8) * CRF_L, 0, 1);

        // B fragments (32x16 f16): N = lane&15; K = (h<8 ? h : 16+(h-8)) + (lane>=16?8:0)
        // -> two runs of 8 consecutive halves per fragment, matching A.
        const int koff = hi * 8;
        union { v16h v; v8h p[2]; } Bh0, Bh1, Bl0, Bl1;
        Bh0.p[0] = *(const v8h*)&phh[n * 72 + koff];
        Bh0.p[1] = *(const v8h*)&phh[n * 72 + koff + 16];
        Bl0.p[0] = *(const v8h*)&pll[n * 72 + koff];
        Bl0.p[1] = *(const v8h*)&pll[n * 72 + koff + 16];
        Bh1.p[0] = *(const v8h*)&phh[n * 72 + 32 + koff];
        Bh1.p[1] = *(const v8h*)&phh[n * 72 + 32 + koff + 16];
        Bl1.p[0] = *(const v8h*)&pll[n * 72 + 32 + koff];
        Bl1.p[1] = *(const v8h*)&pll[n * 72 + 32 + koff + 16];

        v8f c = {};
        c = __builtin_amdgcn_wmma_f32_16x16x32_f16(false, Eh0, false, Bh0.v, (short)0, c, false, false);
        c = __builtin_amdgcn_wmma_f32_16x16x32_f16(false, Eh1, false, Bh1.v, (short)0, c, false, false);
        c = __builtin_amdgcn_wmma_f32_16x16x32_f16(false, Eh0, false, Bl0.v, (short)0, c, false, false);
        c = __builtin_amdgcn_wmma_f32_16x16x32_f16(false, Eh1, false, Bl1.v, (short)0, c, false, false);
        c = __builtin_amdgcn_wmma_f32_16x16x32_f16(false, El0, false, Bh0.v, (short)0, c, false, false);
        c = __builtin_amdgcn_wmma_f32_16x16x32_f16(false, El1, false, Bh1.v, (short)0, c, false, false);

        float mb2 = mrow[n];
        float emv[8] = {c0.x, c0.y, c0.z, c0.w, c1.x, c1.y, c1.z, c1.w};
        float4 w0, w1;
        float na[8];
        #pragma unroll
        for (int q = 0; q < 8; ++q) {
            // raw v_log_f32 is log2; s in {0} U [0.9, 70]; log2(0) = -inf (correct)
            float a    = emv[q] + mb2 + LN2 * __builtin_amdgcn_logf(c[q]);
            float olda = alpha[n * 68 + jb + q];        // lane-private range; no race
            na[q] = (mt > 0.5f) ? a : olda;
        }
        w0.x = na[0]; w0.y = na[1]; w0.z = na[2]; w0.w = na[3];
        w1.x = na[4]; w1.y = na[5]; w1.z = na[6]; w1.w = na[7];
        *(float4*)&alpha[n * 68 + jb]     = w0;
        *(float4*)&alpha[n * 68 + jb + 4] = w1;
        __syncthreads();
    }

    // ---- final: logsumexp(alpha + trans[:, EOS]) - gold (shuffle reduction) ----
    float4 a0 = *(const float4*)&alpha[bloc * 68 + j0];
    float4 a1 = *(const float4*)&alpha[bloc * 68 + j0 + 4];
    float av[8] = {a0.x, a0.y, a0.z, a0.w, a1.x, a1.y, a1.z, a1.w};
    float la[8];
    float pm = -3.402823466e38f;
    #pragma unroll
    for (int k = 0; k < 8; ++k) {
        la[k] = av[k] + trans[(j0 + k) * CRF_L + T_EOS];
        pm = fmaxf(pm, la[k]);
    }
    pm = fmaxf(pm, __shfl_xor(pm, 1));
    pm = fmaxf(pm, __shfl_xor(pm, 2));
    pm = fmaxf(pm, __shfl_xor(pm, 4));
    float s = 0.0f;
    #pragma unroll
    for (int k = 0; k < 8; ++k) s += __builtin_amdgcn_exp2f((la[k] - pm) * LOG2E);
    s += __shfl_xor(s, 1);
    s += __shfl_xor(s, 2);
    s += __shfl_xor(s, 4);
    if ((tid & 7) == 0)
        out[gb0 + bloc] = pm + LN2 * __builtin_amdgcn_logf(s) - gold[gb0 + bloc];
}

// ---------------------------------------------------------------------------
extern "C" void kernel_launch(void* const* d_in, const int* in_sizes, int n_in,
                              void* d_out, int out_size, void* d_ws, size_t ws_size,
                              hipStream_t stream) {
    const float* em    = (const float*)d_in[0];
    const float* trans = (const float*)d_in[1];
    const int*   tags  = (const int*)d_in[2];
    const float* mask  = (const float*)d_in[3];
    float*       out   = (float*)d_out;

    _Float16* eh   = (_Float16*)d_ws;              // 4096 halves (8 KB)
    _Float16* el   = eh + CRF_L * CRF_L;           // 4096 halves (8 KB)
    float*    gold = (float*)(el + CRF_L * CRF_L); // 512 floats  (2 KB)

    crf_prep<<<16, 256, 0, stream>>>(trans, eh, el);
    crf_gold<<<CRF_B / 8, 256, 0, stream>>>(em, trans, tags, mask, gold);
    crf_forward<<<CRF_B / 16, 128, 0, stream>>>(em, trans, mask, eh, el, gold, out);
}
